// RoutedExpertMLP_16965120819357
// MI455X (gfx1250) — compile-verified, matrix operand order checked by reading, not run on hardware
//
#include <hip/hip_runtime.h>
#include <hip/hip_bf16.h>
#include <math.h>

// ---------------- problem constants ----------------
#define D_MODEL   1024
#define D_FF_PER  512
#define N_EXPERTS 8
#define N_TOKENS  8192      // B*T = 4*2048
#define TILE_M    32
#define KPANEL    128       // K-panel for GEMM1 activation staging

typedef __attribute__((ext_vector_type(16))) __bf16 v16bf;
typedef __attribute__((ext_vector_type(8)))  float  v8f;

// ---------------- CDNA5 async-to-LDS helpers (inline asm, ASYNCcnt path) ----
__device__ __forceinline__ void async_load_b128(unsigned lds_off, const void* gaddr) {
    asm volatile("global_load_async_to_lds_b128 %0, %1, off"
                 :: "v"(lds_off), "v"(gaddr) : "memory");
}
__device__ __forceinline__ void wait_async0() {
    asm volatile("s_wait_asynccnt 0" ::: "memory");
}

__device__ __forceinline__ v16bf load_frag(const __bf16* p) {
    union { v16bf v; uint4 u[2]; } f;
    f.u[0] = *(const uint4*)p;
    f.u[1] = *(const uint4*)(p + 16);
    return f.v;
}

// ---------------- prep kernels ----------------
__global__ void k_zero(float* __restrict__ out, int n, int* __restrict__ counts) {
    int i = blockIdx.x * 256 + threadIdx.x;
    if (i < n) out[i] = 0.0f;
    if (i < N_EXPERTS) counts[i] = 0;
}

// 8 elements per thread: 2x float4 in, 1x b128 (8 bf16) out
__global__ void k_cvt_x(const float* __restrict__ x, __bf16* __restrict__ xbf, int n8) {
    int i = blockIdx.x * 256 + threadIdx.x;
    if (i >= n8) return;
    const float4* xp = (const float4*)x + (size_t)i * 2;
    float4 a = xp[0], b = xp[1];
    __bf16 o[8] = {(__bf16)a.x, (__bf16)a.y, (__bf16)a.z, (__bf16)a.w,
                   (__bf16)b.x, (__bf16)b.y, (__bf16)b.z, (__bf16)b.w};
    *(uint4*)(xbf + (size_t)i * 8) = *(const uint4*)o;
}

// W1[e][d][f] -> W1T[e][f][d] bf16. Lane-consecutive f => coalesced reads;
// each thread writes one 16B chunk of 8 consecutive d.
__global__ void k_t_w1(const float* __restrict__ W1, __bf16* __restrict__ W1T) {
    int i = blockIdx.x * 256 + threadIdx.x;         // 8*128*512 total
    int e = i >> 16;
    int rem = i & 0xFFFF;
    int dblk = rem >> 9;                            // 0..127 (block of 8 d)
    int f = rem & 511;
    const float* src = W1 + ((size_t)((e << 10) + dblk * 8)) * 512 + f;
    __bf16 o[8];
#pragma unroll
    for (int q = 0; q < 8; ++q) o[q] = (__bf16)src[(size_t)q * 512];
    *(uint4*)&W1T[((size_t)((e << 9) + f) << 10) + dblk * 8] = *(const uint4*)o;
}

// W2[e][f][d] -> W2T[e][d][f] bf16. Lane-consecutive d => coalesced reads.
__global__ void k_t_w2(const float* __restrict__ W2, __bf16* __restrict__ W2T) {
    int i = blockIdx.x * 256 + threadIdx.x;         // 8*64*1024 total
    int e = i >> 16;
    int rem = i & 0xFFFF;
    int fblk = rem >> 10;                           // 0..63 (block of 8 f)
    int d = rem & 1023;
    const float* src = W2 + ((size_t)((e << 9) + fblk * 8)) * 1024 + d;
    __bf16 o[8];
#pragma unroll
    for (int q = 0; q < 8; ++q) o[q] = (__bf16)src[(size_t)q * 1024];
    *(uint4*)&W2T[((size_t)((e << 10) + d) << 9) + fblk * 8] = *(const uint4*)o;
}

// ---------------- router: one wave (32 lanes) per token ----------------
// List entries are packed (token<<1)|k, k = which of the token's two experts.
__global__ __launch_bounds__(256)
void k_router(const float* __restrict__ x, const float* __restrict__ rw,
              const float* __restrict__ rb, int* __restrict__ counts,
              int* __restrict__ tokList, float* __restrict__ wtList) {
    const int wave = threadIdx.x >> 5;
    const int lane = threadIdx.x & 31;
    const int t = blockIdx.x * 8 + wave;
    if (t >= N_TOKENS) return;

    float acc[N_EXPERTS];
#pragma unroll
    for (int e = 0; e < N_EXPERTS; ++e) acc[e] = 0.0f;

    const float* xr = x + (size_t)t * D_MODEL;
    for (int d = lane; d < D_MODEL; d += 32) {
        float xv = xr[d];
        const float* wr = rw + d * N_EXPERTS;
#pragma unroll
        for (int e = 0; e < N_EXPERTS; ++e) acc[e] += xv * wr[e];
    }
#pragma unroll
    for (int e = 0; e < N_EXPERTS; ++e)
        for (int off = 16; off > 0; off >>= 1)
            acc[e] += __shfl_xor(acc[e], off, 32);

    if (lane == 0) {
        float lg[N_EXPERTS], p[N_EXPERTS];
        float m = -1e30f;
#pragma unroll
        for (int e = 0; e < N_EXPERTS; ++e) { lg[e] = acc[e] + rb[e]; m = fmaxf(m, lg[e]); }
#pragma unroll
        for (int e = 0; e < N_EXPERTS; ++e) p[e] = __expf(lg[e] - m);
        int i0 = 0;
#pragma unroll
        for (int e = 1; e < N_EXPERTS; ++e) if (p[e] > p[i0]) i0 = e;
        int i1 = (i0 == 0) ? 1 : 0;
#pragma unroll
        for (int e = 0; e < N_EXPERTS; ++e) if (e != i0 && p[e] > p[i1]) i1 = e;
        float inv = 1.0f / (p[i0] + p[i1]);
        int pos0 = atomicAdd(&counts[i0], 1);
        tokList[i0 * N_TOKENS + pos0] = (t << 1);
        wtList[i0 * N_TOKENS + pos0] = p[i0] * inv;
        int pos1 = atomicAdd(&counts[i1], 1);
        tokList[i1 * N_TOKENS + pos1] = (t << 1) | 1;
        wtList[i1 * N_TOKENS + pos1] = p[i1] * inv;
    }
}

// ---------------- fused expert FFN (WMMA bf16, f32 accumulate) ----------------
__global__ __launch_bounds__(256)
void k_moe_ffn(const __bf16* __restrict__ xbf, const __bf16* __restrict__ W1T,
               const __bf16* __restrict__ W2T, const float* __restrict__ b1,
               const float* __restrict__ b2, const int* __restrict__ counts,
               const int* __restrict__ tokList, const float* __restrict__ wtList,
               float* __restrict__ out, float* __restrict__ partial) {
    const int e    = blockIdx.y;
    const int cnt  = counts[e];
    const int base = blockIdx.x * TILE_M;
    if (base >= cnt) return;

    __shared__ __bf16 sPan[2][TILE_M * KPANEL];   // 2 x 8 KB activation panels
    __shared__ __bf16 sH[TILE_M * D_FF_PER];      // 32 KB GELU(h1)
    __shared__ int    sTok[TILE_M];               // gather token index
    __shared__ int    sSlot[TILE_M];              // output slot (branchless pads)
    __shared__ float  sWgt[TILE_M];

    const int tid = threadIdx.x;
    if (tid < TILE_M) {
        int r = base + tid;
        if (r < cnt) {
            int entry = tokList[e * N_TOKENS + r];
            sTok[tid]  = entry >> 1;
            sSlot[tid] = entry;
            sWgt[tid]  = wtList[e * N_TOKENS + r];
        } else {
            // pad row: gather a valid token; output goes to dump slot (partial
            // path) or is a weight-0 add to a valid token (atomic path).
            int entry = tokList[e * N_TOKENS + base];
            sTok[tid]  = entry >> 1;
            sSlot[tid] = partial ? (2 * N_TOKENS) : entry;
            sWgt[tid]  = 0.0f;
        }
    }
    __syncthreads();

    const int wave = tid >> 5;
    const int lane = tid & 31;
    const int half = (lane >> 4) << 3;   // 0 or 8: per-lane K sub-chunk base
    const int rc   = lane & 15;          // A row / B column within tile

    // per-thread staging coords: 512 x 16B chunks per panel, 2 per thread
    const int c0row = tid >> 4, c0col = (tid & 15) << 3;
    const int c1row = (tid + 256) >> 4, c1col = c0col;
    const unsigned ldsPan0 = (unsigned)(size_t)&sPan[0][0];
    const unsigned ldsPan1 = (unsigned)(size_t)&sPan[1][0];

    // issue panel 0
    async_load_b128(ldsPan0 + (unsigned)(c0row * KPANEL + c0col) * 2,
                    xbf + (size_t)sTok[c0row] * D_MODEL + c0col);
    async_load_b128(ldsPan0 + (unsigned)(c1row * KPANEL + c1col) * 2,
                    xbf + (size_t)sTok[c1row] * D_MODEL + c1col);

    // ---- GEMM1: x[32x1024] @ W1T -> GELU -> sH[32x512]
    v8f acc1[2][4] = {};
    const int n0w1 = wave * 64;                       // 4 N-tiles per wave
    const __bf16* w1base = W1T + ((size_t)(e * D_FF_PER + n0w1 + rc) << 10);

    for (int P = 0; P < D_MODEL / KPANEL; ++P) {
        wait_async0();
        __syncthreads();                               // panel P resident in LDS
        if (P + 1 < D_MODEL / KPANEL) {                // prefetch panel P+1
            const int kn = (P + 1) * KPANEL;
            unsigned dst = (P & 1) ? ldsPan0 : ldsPan1;
            async_load_b128(dst + (unsigned)(c0row * KPANEL + c0col) * 2,
                            xbf + (size_t)sTok[c0row] * D_MODEL + kn + c0col);
            async_load_b128(dst + (unsigned)(c1row * KPANEL + c1col) * 2,
                            xbf + (size_t)sTok[c1row] * D_MODEL + kn + c1col);
        }
        const __bf16* pan = &sPan[P & 1][0];
        const int kbase = P * KPANEL;
#pragma unroll
        for (int kk = 0; kk < KPANEL / 32; ++kk) {
            const int k0 = kk * 32;
            v16bf af0 = load_frag(pan + (rc)      * KPANEL + k0 + half);
            v16bf af1 = load_frag(pan + (16 + rc) * KPANEL + k0 + half);
#pragma unroll
            for (int j = 0; j < 4; ++j) {
                v16bf bf = load_frag(w1base + ((size_t)(j * 16) << 10) + kbase + k0 + half);
                acc1[0][j] = __builtin_amdgcn_wmma_f32_16x16x32_bf16(
                    false, af0, false, bf, (short)0, acc1[0][j], false, false);
                acc1[1][j] = __builtin_amdgcn_wmma_f32_16x16x32_bf16(
                    false, af1, false, bf, (short)0, acc1[1][j], false, false);
            }
        }
    }
    __syncthreads();                                    // all reads of panels done
#pragma unroll
    for (int mt = 0; mt < 2; ++mt)
#pragma unroll
        for (int j = 0; j < 4; ++j) {
            int n = n0w1 + j * 16 + rc;
            float bias = b1[e * D_FF_PER + n];
#pragma unroll
            for (int r = 0; r < 8; ++r) {
                int m = mt * 16 + r + half;             // C/D layout: lanes>=16 -> M+8
                float v = acc1[mt][j][r] + bias;
                float g = 0.5f * v * (1.0f + erff(v * 0.70710678118654752f));
                sH[m * D_FF_PER + n] = (__bf16)g;
            }
        }
    __syncthreads();

    // ---- GEMM2: sH[32x512] @ W2T -> gate-scaled output
    v8f acc2[2][8] = {};
    const int n0w2 = wave * 128;                       // 8 N-tiles per wave
    const __bf16* w2base = W2T + ((size_t)(e * D_MODEL + n0w2 + rc) << 9);
    for (int k0 = 0; k0 < D_FF_PER; k0 += 32) {
        v16bf af0 = load_frag(sH + (rc)      * D_FF_PER + k0 + half);
        v16bf af1 = load_frag(sH + (16 + rc) * D_FF_PER + k0 + half);
#pragma unroll
        for (int j = 0; j < 8; ++j) {
            v16bf bf = load_frag(w2base + ((size_t)(j * 16) << 9) + k0 + half);
            acc2[0][j] = __builtin_amdgcn_wmma_f32_16x16x32_bf16(
                false, af0, false, bf, (short)0, acc2[0][j], false, false);
            acc2[1][j] = __builtin_amdgcn_wmma_f32_16x16x32_bf16(
                false, af1, false, bf, (short)0, acc2[1][j], false, false);
        }
    }

    // branchless epilogues; single uniform branch on output mode
    if (partial) {
#pragma unroll
        for (int mt = 0; mt < 2; ++mt)
#pragma unroll
            for (int j = 0; j < 8; ++j) {
                int n = n0w2 + j * 16 + rc;
                float bias = b2[e * D_MODEL + n];
#pragma unroll
                for (int r = 0; r < 8; ++r) {
                    int m = mt * 16 + r + half;
                    float v = (acc2[mt][j][r] + bias) * sWgt[m];
                    partial[((size_t)sSlot[m] << 10) + n] = v;
                }
            }
    } else {
#pragma unroll
        for (int mt = 0; mt < 2; ++mt)
#pragma unroll
            for (int j = 0; j < 8; ++j) {
                int n = n0w2 + j * 16 + rc;
                float bias = b2[e * D_MODEL + n];
#pragma unroll
                for (int r = 0; r < 8; ++r) {
                    int m = mt * 16 + r + half;
                    float v = (acc2[mt][j][r] + bias) * sWgt[m];
                    atomicAdd(&out[((size_t)(sSlot[m] >> 1) << 10) + n], v);
                }
            }
    }
}

// out[t][d] = partial[2t][d] + partial[2t+1][d], float4-vectorized
__global__ void k_reduce(const float4* __restrict__ partial, float4* __restrict__ out, int n4) {
    int i = blockIdx.x * 256 + threadIdx.x;
    if (i >= n4) return;
    int t = i >> 8;                                    // 256 float4 per token row
    int d4 = i & 255;
    float4 a = partial[((size_t)(t << 1) << 8) + d4];
    float4 b = partial[((size_t)((t << 1) | 1) << 8) + d4];
    float4 o; o.x = a.x + b.x; o.y = a.y + b.y; o.z = a.z + b.z; o.w = a.w + b.w;
    out[i] = o;
}

// ---------------- host launch ----------------
extern "C" void kernel_launch(void* const* d_in, const int* in_sizes, int n_in,
                              void* d_out, int out_size, void* d_ws, size_t ws_size,
                              hipStream_t stream) {
    const float* x  = (const float*)d_in[0];
    const float* rw = (const float*)d_in[1];
    const float* rb = (const float*)d_in[2];
    const float* W1 = (const float*)d_in[3];
    const float* b1 = (const float*)d_in[4];
    const float* W2 = (const float*)d_in[5];
    const float* b2 = (const float*)d_in[6];
    float* out = (float*)d_out;

    char* ws = (char*)d_ws;
    size_t off = 0;
    __bf16* xbf = (__bf16*)(ws + off); off += (size_t)N_TOKENS * D_MODEL * 2;
    __bf16* W1T = (__bf16*)(ws + off); off += (size_t)N_EXPERTS * D_FF_PER * D_MODEL * 2;
    __bf16* W2T = (__bf16*)(ws + off); off += (size_t)N_EXPERTS * D_MODEL * D_FF_PER * 2;
    int*   counts  = (int*)(ws + off); off += 256;
    int*   tokList = (int*)(ws + off); off += (size_t)N_EXPERTS * N_TOKENS * 4;
    float* wtList  = (float*)(ws + off); off += (size_t)N_EXPERTS * N_TOKENS * 4;
    size_t partBytes = (size_t)(2 * N_TOKENS + 1) * D_MODEL * 4;   // +1 dump slot
    float* partial = (ws_size >= off + partBytes) ? (float*)(ws + off) : nullptr;
    (void)in_sizes; (void)n_in; (void)out_size;

    const int nOut = N_TOKENS * D_MODEL;             // 8388608
    const int nT   = 524288;                         // transpose threads (x2 kernels)

    if (partial) k_zero<<<dim3(1), dim3(256), 0, stream>>>(out, 0, counts);
    else         k_zero<<<dim3((nOut + 255) / 256), dim3(256), 0, stream>>>(out, nOut, counts);
    k_cvt_x<<<dim3((nOut / 8 + 255) / 256), dim3(256), 0, stream>>>(x, xbf, nOut / 8);
    k_t_w1<<<dim3(nT / 256), dim3(256), 0, stream>>>(W1, W1T);
    k_t_w2<<<dim3(nT / 256), dim3(256), 0, stream>>>(W2, W2T);
    k_router<<<dim3(N_TOKENS / 8), dim3(256), 0, stream>>>(x, rw, rb, counts, tokList, wtList);
    k_moe_ffn<<<dim3(N_TOKENS / TILE_M, N_EXPERTS), dim3(256), 0, stream>>>(
        xbf, W1T, W2T, b1, b2, counts, tokList, wtList, out, partial);
    if (partial)
        k_reduce<<<dim3((nOut / 4 + 255) / 256), dim3(256), 0, stream>>>(
            (const float4*)partial, (float4*)out, nOut / 4);
}
